// GeometricLinear_2834678415614
// MI455X (gfx1250) — matrix operand order repaired and checked by reading, compiled, not run on hardware
//
#include <hip/hip_runtime.h>
#include <math.h>

typedef __attribute__((ext_vector_type(16))) _Float16 v16h;
typedef __attribute__((ext_vector_type(8)))  _Float16 v8h;
typedef __attribute__((ext_vector_type(8)))  float    v8f;
typedef __attribute__((ext_vector_type(4)))  int      v4i;

#define IN_F   64
#define OUT_F  64
#define KDIM   2048   // IN_F * 32
#define NDIM   2048   // OUT_F * 32
#define BM     128
#define BN     128
#define BKSTEP 64     // K elements staged per iteration (2 WMMA k-steps)
#define BKPAD  72     // padded LDS row stride (halves): 144B -> conflict-free
#define BUFSZ  (BM * BKPAD)

// Async global->LDS path (gfx1250); guaranteed fallback keeps compile green.
#if defined(__HIP_DEVICE_COMPILE__) && \
    __has_builtin(__builtin_amdgcn_global_load_async_to_lds_b128) && \
    __has_builtin(__builtin_amdgcn_s_wait_asynccnt)
#define USE_ASYNC 1
typedef __attribute__((address_space(1))) v4i GINT4;  // global int4
typedef __attribute__((address_space(3))) v4i LINT4;  // LDS int4
#else
#define USE_ASYNC 0
#endif

// ---------------------------------------------------------------------------
// Cl(4,1) geometric-product sign: sign of e_a * e_b (result blade = a ^ b)
// ---------------------------------------------------------------------------
__device__ __forceinline__ float gp_sign(int a, int b) {
  float s = 1.0f;
  int bits = a;
  #pragma unroll
  for (int i = 0; i < 5; ++i) {
    if ((b >> i) & 1) {
      int high = bits >> (i + 1);           // generators of `bits` above i
      if (__popc(high) & 1) s = -s;         // anticommutation swaps
      if ((bits >> i) & 1) {
        if (i == 4) s = -s;                 // e4^2 = -1 (signature 4,1)
        bits &= ~(1 << i);
      } else {
        bits |= (1 << i);
      }
    }
  }
  return s;
}

// ---------------------------------------------------------------------------
// Kernel 1: convert x (f32) -> f16, 8 elements per thread
// ---------------------------------------------------------------------------
__global__ void cvt_x_f16(const float* __restrict__ x, _Float16* __restrict__ xh,
                          int n8) {
  int t = blockIdx.x * blockDim.x + threadIdx.x;
  if (t >= n8) return;
  const float4* p = (const float4*)x + (size_t)t * 2;
  float4 f0 = p[0], f1 = p[1];
  v8h h;
  h[0] = (_Float16)f0.x; h[1] = (_Float16)f0.y;
  h[2] = (_Float16)f0.z; h[3] = (_Float16)f0.w;
  h[4] = (_Float16)f1.x; h[5] = (_Float16)f1.y;
  h[6] = (_Float16)f1.z; h[7] = (_Float16)f1.w;
  ((v8h*)xh)[t] = h;
}

// ---------------------------------------------------------------------------
// Kernel 2: effective weight, K-major (acts as B^T for the GEMM)
//   Wt[col=o*32+kb][red=i*32+l] = gp_sign(kb^l, l) * weight[o,i,kb^l]
// ---------------------------------------------------------------------------
__global__ void build_wt(const float* __restrict__ w, _Float16* __restrict__ wt) {
  int t = blockIdx.x * blockDim.x + threadIdx.x;   // 0 .. NDIM*KDIM-1
  int red = t & (KDIM - 1);
  int col = t >> 11;                               // KDIM = 2^11
  int o  = col >> 5, kb = col & 31;
  int i  = red >> 5, l  = red & 31;
  int j  = kb ^ l;                                 // GP[j,l,kb] nonzero
  float s = gp_sign(j, l);
  wt[t] = (_Float16)(s * w[(o * IN_F + i) * 32 + j]);
}

// ---------------------------------------------------------------------------
// Kernel 3: WMMA GEMM  out[M,N] = Xh[M,K] * Wt[N,K]^T   (f16 in, f32 acc)
// 256 threads (8 waves), tile 128x128x64; waves 4(M) x 2(N) -> 32x64 each.
// Async double-buffered global->LDS when available.
// ---------------------------------------------------------------------------
__global__ __launch_bounds__(256)
void gemm_wmma(const _Float16* __restrict__ Xh, const _Float16* __restrict__ Wt,
               float* __restrict__ out) {
#if USE_ASYNC
  __shared__ _Float16 Xs[2 * BUFSZ];
  __shared__ _Float16 Ws[2 * BUFSZ];
#else
  __shared__ _Float16 Xs[BUFSZ];
  __shared__ _Float16 Ws[BUFSZ];
#endif

  const int tid   = threadIdx.x;
  const int lane  = tid & 31;
  const int wave  = tid >> 5;
  const int waveM = wave & 3;    // *32 in M
  const int waveN = wave >> 2;   // *64 in N
  const int bM = blockIdx.y * BM;
  const int bN = blockIdx.x * BN;

  // staging: thread covers rows tid/8 + {0,32,64,96}, col chunk (tid&7)*8 halves
  const int grow = tid >> 3;
  const int gcol = (tid & 7) * 8;
  const _Float16* gx = Xh + (size_t)(bM + grow) * KDIM + gcol;
  const _Float16* gw = Wt + (size_t)(bN + grow) * KDIM + gcol;
  const int lds_off = grow * BKPAD + gcol;

  const int ml = lane & 15;      // row within 16x16 fragment
  const int kh = lane >> 4;      // K-half selector

  v8f acc[2][4] = {};
  const int NT = KDIM / BKSTEP;  // 32 iterations

#if USE_ASYNC
  // prologue: issue tile 0 into buffer 0 (8 async b128 per thread)
  #pragma unroll
  for (int rr = 0; rr < 4; ++rr) {
    __builtin_amdgcn_global_load_async_to_lds_b128(
        (GINT4*)(gx + (size_t)rr * 32 * KDIM),
        (LINT4*)&Xs[lds_off + rr * 32 * BKPAD], 0, 0);
    __builtin_amdgcn_global_load_async_to_lds_b128(
        (GINT4*)(gw + (size_t)rr * 32 * KDIM),
        (LINT4*)&Ws[lds_off + rr * 32 * BKPAD], 0, 0);
  }
#endif

  for (int it = 0; it < NT; ++it) {
#if USE_ASYNC
    const int c  = it & 1;
    const int cb = c * BUFSZ;
    if (it + 1 < NT) {
      const int nb = (1 - c) * BUFSZ;
      const size_t k0 = (size_t)(it + 1) * BKSTEP;
      #pragma unroll
      for (int rr = 0; rr < 4; ++rr) {
        __builtin_amdgcn_global_load_async_to_lds_b128(
            (GINT4*)(gx + k0 + (size_t)rr * 32 * KDIM),
            (LINT4*)&Xs[nb + lds_off + rr * 32 * BKPAD], 0, 0);
        __builtin_amdgcn_global_load_async_to_lds_b128(
            (GINT4*)(gw + k0 + (size_t)rr * 32 * KDIM),
            (LINT4*)&Ws[nb + lds_off + rr * 32 * BKPAD], 0, 0);
      }
      __builtin_amdgcn_s_wait_asynccnt(8);   // current tile's 8 loads done
    } else {
      __builtin_amdgcn_s_wait_asynccnt(0);
    }
    __syncthreads();                          // all waves' loads landed
#else
    const int cb = 0;
    const size_t k0 = (size_t)it * BKSTEP;
    v8h xr[4], wr[4];
    #pragma unroll
    for (int rr = 0; rr < 4; ++rr) {
      xr[rr] = *(const v8h*)(gx + k0 + (size_t)rr * 32 * KDIM);
      wr[rr] = *(const v8h*)(gw + k0 + (size_t)rr * 32 * KDIM);
    }
    __syncthreads();                          // previous tile consumed
    #pragma unroll
    for (int rr = 0; rr < 4; ++rr) {
      *(v8h*)&Xs[lds_off + rr * 32 * BKPAD] = xr[rr];
      *(v8h*)&Ws[lds_off + rr * 32 * BKPAD] = wr[rr];
    }
    __syncthreads();                          // tile visible
#endif

    // ---- compute: 2 k-substeps x 8 WMMAs on buffer cb ----
    #pragma unroll
    for (int ks = 0; ks < BKSTEP; ks += 32) {
      v16h afrag[2];
      #pragma unroll
      for (int am = 0; am < 2; ++am) {
        int r = waveM * 32 + am * 16 + ml;
        const int base = cb + r * BKPAD + ks;
        v8h lo = *(const v8h*)&Xs[base + kh * 8];
        v8h hi = *(const v8h*)&Xs[base + 16 + kh * 8];
        afrag[am] = __builtin_shufflevector(lo, hi,
                      0,1,2,3,4,5,6,7,8,9,10,11,12,13,14,15);
      }
      v16h bfrag[4];
      #pragma unroll
      for (int bn = 0; bn < 4; ++bn) {
        int r = waveN * 64 + bn * 16 + ml;
        const int base = cb + r * BKPAD + ks;
        v8h lo = *(const v8h*)&Ws[base + kh * 8];
        v8h hi = *(const v8h*)&Ws[base + 16 + kh * 8];
        bfrag[bn] = __builtin_shufflevector(lo, hi,
                      0,1,2,3,4,5,6,7,8,9,10,11,12,13,14,15);
      }
      #pragma unroll
      for (int am = 0; am < 2; ++am)
        #pragma unroll
        for (int bn = 0; bn < 4; ++bn)
          acc[am][bn] = __builtin_amdgcn_wmma_f32_16x16x32_f16(
              false, afrag[am], false, bfrag[bn],
              (short)0, acc[am][bn], false, false);
    }

#if USE_ASYNC
    __syncthreads();   // buffer cb free before iteration it+1 re-issues into it
#endif
  }

  // ---- epilogue: C layout VGPR r -> M = r (lanes 0-15) / r+8 (lanes 16-31) ----
  #pragma unroll
  for (int am = 0; am < 2; ++am)
    #pragma unroll
    for (int bn = 0; bn < 4; ++bn) {
      int col = bN + waveN * 64 + bn * 16 + ml;
      #pragma unroll
      for (int r = 0; r < 8; ++r) {
        int row = bM + waveM * 32 + am * 16 + kh * 8 + r;
        out[(size_t)row * NDIM + col] = acc[am][bn][r];
      }
    }
}

// ---------------------------------------------------------------------------
// Kernel 4: manifold normalization, in place; one thread per (b,s,o) row of 32
// ---------------------------------------------------------------------------
__global__ void mnorm(float* __restrict__ out, int nrows) {
  int t = blockIdx.x * blockDim.x + threadIdx.x;
  if (t >= nrows) return;
  float4* q = (float4*)(out + (size_t)t * 32);
  float v[32];
  #pragma unroll
  for (int i = 0; i < 8; ++i) {
    float4 f = q[i];
    v[4*i] = f.x; v[4*i+1] = f.y; v[4*i+2] = f.z; v[4*i+3] = f.w;
  }
  float ns = 0.f, l2 = 0.f;
  #pragma unroll
  for (int k = 0; k < 32; ++k) {
    int g = __popc(k);
    float sig = 1.f;
    if (k & 16) sig = -sig;                  // e4 in blade -> metric -1
    if (((g * (g - 1)) / 2) & 1) sig = -sig; // reversal sign
    float vv = v[k] * v[k];
    ns += vv * sig;
    l2 += vv;
  }
  float a  = sqrtf(fabsf(ns) + 1e-8f);
  float bb = sqrtf(l2) * 0.25f + 1e-8f;
  float inv = 1.f / fmaxf(fmaxf(a, bb), 1.f);
  #pragma unroll
  for (int i = 0; i < 8; ++i) {
    float4 f = q[i];
    f.x *= inv; f.y *= inv; f.z *= inv; f.w *= inv;
    q[i] = f;
  }
}

// ---------------------------------------------------------------------------
extern "C" void kernel_launch(void* const* d_in, const int* in_sizes, int n_in,
                              void* d_out, int out_size, void* d_ws, size_t ws_size,
                              hipStream_t stream) {
  const float* x = (const float*)d_in[0];   // [B,S,IN,32] f32
  const float* w = (const float*)d_in[1];   // [OUT,IN,32] f32
  float* out = (float*)d_out;               // [B,S,OUT,32] f32

  const int M = in_sizes[0] / KDIM;         // B*S = 8192

  _Float16* xh = (_Float16*)d_ws;                                   // M*KDIM halves
  _Float16* wt = (_Float16*)((char*)d_ws + (size_t)M * KDIM * 2);   // NDIM*KDIM halves

  // 1) x -> f16
  int n8 = in_sizes[0] / 8;
  cvt_x_f16<<<(n8 + 255) / 256, 256, 0, stream>>>(x, xh, n8);

  // 2) effective weight (B^T), f16
  build_wt<<<(NDIM * KDIM) / 256, 256, 0, stream>>>(w, wt);

  // 3) WMMA GEMM -> raw f32 output
  dim3 grid(NDIM / BN, M / BM);
  gemm_wmma<<<grid, 256, 0, stream>>>(xh, wt, out);

  // 4) manifold normalization in place
  int nrows = M * OUT_F;
  mnorm<<<(nrows + 255) / 256, 256, 0, stream>>>(out, nrows);
}
